// Mamba3Baseline_17669495456482
// MI455X (gfx1250) — matrix operand order repaired and checked by reading, not compiled, not run on hardware
//
#include <hip/hip_runtime.h>
#include <stdint.h>

// Mamba3 chunked causal decay-attention for MI455X (gfx1250, wave32, WMMA,
// async global->LDS double buffering).
// b=2, l=2048, h=8, dh=64, g=1, n=64, chunk=64 (hardcoded from the reference).

#define BB 2
#define LL 2048
#define HH 8
#define DH 64
#define CH 64

typedef __attribute__((ext_vector_type(16))) _Float16 v16h;
typedef __attribute__((ext_vector_type(8)))  float    v8f;
typedef __attribute__((ext_vector_type(8)))  uint32_t v8u;

__device__ __forceinline__ v8f zero8() {
    v8f z = {0.f, 0.f, 0.f, 0.f, 0.f, 0.f, 0.f, 0.f};
    return z;
}

// Low 32 bits of the generic (flat) address of an LDS object == LDS byte
// offset.
__device__ __forceinline__ uint32_t lds_off(const void* p) {
    return (uint32_t)(uintptr_t)p;
}

// ---------------------------------------------------------------------------
// A-matrix fragment (16x32 f16, M x K), read from row-major [m][k] LDS tile.
// lanes 0-15 -> M=lane; VGPR j<4 holds K pair (half*8 + 2j), VGPR j>=4 holds
// K pair (16 + half*8 + 2(j-4)); lanes 16-31 are half=1. Pairs are
// K-contiguous and 4B aligned -> b32 LDS reads (compiler merges to b128).
// ---------------------------------------------------------------------------
__device__ __forceinline__ v16h load_a16(const _Float16* base, int ld, int lane) {
    const int m  = lane & 15;
    const int hl = (lane >> 4) & 1;
    const uint32_t* row = (const uint32_t*)(base + m * ld);
    v8u u;
#pragma unroll
    for (int j = 0; j < 8; ++j) {
        const int k0 = (j < 4) ? (hl * 8 + 2 * j) : (16 + hl * 8 + 2 * (j - 4));
        u[j] = row[k0 >> 1];
    }
    return __builtin_bit_cast(v16h, u);
}

// ---------------------------------------------------------------------------
// B-matrix fragment (32x16 f16, K x N), read from row-major [n][k] LDS tile
// (tile stored transposed so K pairs are contiguous).
// lanes 0-15 -> N=lane; VGPR j<4 holds K pair (half*8 + 2j), VGPR j>=4 holds
// K pair (16 + half*8 + 2(j-4)); lanes 16-31 are half=1 (N=lane-16).
// ---------------------------------------------------------------------------
__device__ __forceinline__ v16h load_bT16(const _Float16* base, int ld, int lane) {
    const int nn = lane & 15;
    const int hl = (lane >> 4) & 1;
    const uint32_t* row = (const uint32_t*)(base + nn * ld);
    v8u u;
#pragma unroll
    for (int j = 0; j < 8; ++j) {
        const int k0 = (j < 4) ? (hl * 8 + 2 * j) : (16 + hl * 8 + 2 * (j - 4));
        u[j] = row[k0 >> 1];
    }
    return __builtin_bit_cast(v16h, u);
}

__device__ __forceinline__ v8f wmma_f16(v16h a, v16h b, v8f c) {
    return __builtin_amdgcn_wmma_f32_16x16x32_f16(
        /*neg_a=*/false, a, /*neg_b=*/false, b,
        /*c_mod=*/(short)0, c, /*reuse_a=*/false, /*reuse_b=*/false);
}

// ---------------------------------------------------------------------------
// Async global->LDS staging of one key chunk (raw f32): B tile is a
// contiguous 16 KB region; x tile is 64 rows x 256 B at 2 KB stride.
// 128 threads x 8 issues x 16 B = 16 KB per tile. Tracked by ASYNCcnt.
// ---------------------------------------------------------------------------
__device__ __forceinline__ void async_load_chunk(const float* gB, const float* gX,
                                                 float* rawB, float* rawX, int tid) {
    const uint32_t lB = lds_off(rawB);
    const uint32_t lX = lds_off(rawX);
    const uint64_t aB = (uint64_t)(uintptr_t)gB;
    const uint64_t aX = (uint64_t)(uintptr_t)gX;
#pragma unroll
    for (int i = 0; i < 8; ++i) {
        const uint32_t u = (uint32_t)(i * 128 + tid);      // 16B unit index
        // B chunk: fully contiguous.
        {
            const uint32_t la = lB + u * 16u;
            const uint64_t ga = aB + (uint64_t)u * 16u;
            asm volatile("global_load_async_to_lds_b128 %0, %1, off"
                         :: "v"(la), "v"(ga) : "memory");
        }
        // x chunk: row (u>>4) at stride HH*DH*4 bytes, 16 units per row.
        {
            const uint32_t la = lX + u * 16u;
            const uint64_t ga = aX + (uint64_t)(u >> 4) * (HH * DH * 4)
                                   + (uint64_t)(u & 15) * 16u;
            asm volatile("global_load_async_to_lds_b128 %0, %1, off"
                         :: "v"(la), "v"(ga) : "memory");
        }
    }
}

// ===========================================================================
// Pass 1: ang[bh][l] = inclusive cumsum over l of dt[b,l,h].
// (cs = A[h]*ang, so one scan serves both the RoPE angle and the decay.)
// ===========================================================================
__global__ __launch_bounds__(256)
void mamba3_scan_kernel(const float* __restrict__ dt, float* __restrict__ ang) {
    const int bh = blockIdx.x;            // 0..15
    const int b  = bh >> 3;
    const int h  = bh & 7;
    const int t  = threadIdx.x;           // 0..255

    float vals[8];
    float run = 0.f;
#pragma unroll
    for (int i = 0; i < 8; ++i) {
        const int l = t * 8 + i;
        vals[i] = dt[(b * LL + l) * HH + h];
        run += vals[i];
    }

    __shared__ float sums[256];
    sums[t] = run;
    __syncthreads();
#pragma unroll
    for (int off = 1; off < 256; off <<= 1) {
        const float v = (t >= off) ? sums[t - off] : 0.f;
        __syncthreads();
        sums[t] += v;
        __syncthreads();
    }
    float acc = sums[t] - run;            // exclusive prefix for this thread
#pragma unroll
    for (int i = 0; i < 8; ++i) {
        acc += vals[i];
        ang[bh * LL + t * 8 + i] = acc;   // inclusive cumsum
    }
}

// ===========================================================================
// Pass 2: fused chunked causal decay-attention.
// grid = (b*h) * (l/64) blocks, 128 threads = 4 waves. Wave w owns the
// 16-row output strip. Per key chunk kj <= qi:
//   raw B/x tiles arrive via async global->LDS (double buffered, prefetched
//   one chunk ahead, drained with s_wait_asynccnt 0)
//   S(16x64) = Qrope * Krope^T        (8x v_wmma_f32_16x16x32_f16)
//   P = S * exp(csq - csk) * w_s * causal-mask   (registers)
//   Y(16x64) += P * V                 (8x v_wmma_f32_16x16x32_f16)
// ===========================================================================
__global__ __launch_bounds__(128)
void mamba3_attn_kernel(const float* __restrict__ x,
                        const float* __restrict__ dt,
                        const float* __restrict__ A,
                        const float* __restrict__ Bp,
                        const float* __restrict__ Cp,
                        const float* __restrict__ ang,
                        float* __restrict__ out) {
    __shared__ _Float16 Qs[CH * DH];   // RoPE'd Q, row-major [m][d]
    __shared__ _Float16 Ks[CH * DH];   // RoPE'd K, row-major [s][d]
    __shared__ _Float16 Vt[DH * CH];   // V transposed, [d][s]
    __shared__ _Float16 Ps[CH * CH];   // decayed scores, wave-private strips
    __shared__ float    csq[CH], csk[CH], wk[CH];
    alignas(16) __shared__ float Braw[2][CH * DH];   // async staging (f32)
    alignas(16) __shared__ float Xraw[2][CH * DH];   // async staging (f32)

    const int qi   = blockIdx.x & 31;
    const int bh   = blockIdx.x >> 5;
    const int b    = bh >> 3;
    const int h    = bh & 7;
    const int tid  = threadIdx.x;
    const int lane = tid & 31;
    const int wave = tid >> 5;          // 0..3
    const int l0q  = qi * CH;
    const float Ah = A[h];

    // Kick off async staging of key chunk 0 before anything else.
    async_load_chunk(Bp + (size_t)(b * LL) * 64,
                     x + ((size_t)(b * LL) * HH + h) * DH,
                     Braw[0], Xraw[0], tid);

    // ---- stage RoPE'd Q tile + csq (two threads per row, 32 dims each) ----
    {
        const int r  = tid >> 1;                 // 0..63
        const int d0 = (tid & 1) * 32;
        const int lq = l0q + r;
        const float a = ang[bh * LL + lq];
        float sn, cn;
        __sincosf(a, &sn, &cn);
        const float* crow = Cp + (b * LL + lq) * 64;   // g=1, n=64
#pragma unroll 8
        for (int i = 0; i < 32; ++i) {
            const int d = d0 + i;
            const float v = (d < 32) ? (crow[d] * cn - crow[d + 32] * sn)
                                     : (crow[d - 32] * sn + crow[d] * cn);
            Qs[r * DH + d] = (_Float16)v;
        }
        if ((tid & 1) == 0) csq[r] = Ah * a;
    }
    __syncthreads();

    // Q A-fragments for this wave's strip (hoisted out of the kj loop).
    const v16h qa0 = load_a16(Qs + wave * 16 * DH + 0,  DH, lane);
    const v16h qa1 = load_a16(Qs + wave * 16 * DH + 32, DH, lane);

    v8f yacc[4];
#pragma unroll
    for (int t = 0; t < 4; ++t) yacc[t] = zero8();

    for (int kj = 0; kj <= qi; ++kj) {
        const int l0k = kj * CH;
        const int buf = kj & 1;

        // Drain this wave's async issues, then rendezvous so the whole raw
        // tile (written by all waves) is visible; also fences Ks/Vt reuse.
        asm volatile("s_wait_asynccnt 0" ::: "memory");
        __syncthreads();

        // ---- rope-convert raw[buf] -> Ks / Vt, per-key decay scalars ----
        {
            const int s  = tid >> 1;
            const int d0 = (tid & 1) * 32;
            const int ls = l0k + s;
            const float a = ang[bh * LL + ls];
            float sn, cn;
            __sincosf(a, &sn, &cn);
            const float* brow = &Braw[buf][s * DH];
            const float* xrow = &Xraw[buf][s * DH];
#pragma unroll 8
            for (int i = 0; i < 32; ++i) {
                const int d = d0 + i;
                const float kv = (d < 32) ? (brow[d] * cn - brow[d + 32] * sn)
                                          : (brow[d - 32] * sn + brow[d] * cn);
                Ks[s * DH + d] = (_Float16)kv;
                Vt[d * CH + s] = (_Float16)xrow[d];
            }
            if ((tid & 1) == 0) {
                csk[s] = Ah * a;
                const float adtv = Ah * dt[(b * LL + ls) * HH + h];
                wk[s] = 0.5f * (1.f + __expf(adtv));
            }
        }

        // Prefetch next key chunk into the other buffer (overlaps compute).
        if (kj < qi) {
            const int ln = l0k + CH;
            async_load_chunk(Bp + ((size_t)(b * LL + ln)) * 64,
                             x + (((size_t)(b * LL + ln)) * HH + h) * DH,
                             Braw[buf ^ 1], Xraw[buf ^ 1], tid);
        }
        __syncthreads();

        // ---- S = Q K^T : 4 N-tiles x 2 K-steps ----
        v8f sacc[4];
#pragma unroll
        for (int t = 0; t < 4; ++t) {
            sacc[t] = zero8();
            const v16h b0 = load_bT16(Ks + t * 16 * DH + 0,  DH, lane);
            sacc[t] = wmma_f16(qa0, b0, sacc[t]);
            const v16h b1 = load_bT16(Ks + t * 16 * DH + 32, DH, lane);
            sacc[t] = wmma_f16(qa1, b1, sacc[t]);
        }

        // ---- decay * mask, write P strip (wave-private) ----
        {
            const int nn = lane & 15;
            const int hl = (lane >> 4) & 1;
            _Float16* Pw = Ps + wave * 16 * CH;
            const bool diag = (kj == qi);
#pragma unroll
            for (int t = 0; t < 4; ++t) {
                const int scol = t * 16 + nn;
                const float ck = csk[scol];
                const float wv = wk[scol];
#pragma unroll
                for (int e = 0; e < 8; ++e) {
                    const int m = e + 8 * hl;          // row within strip
                    const int mrow = wave * 16 + m;    // row within chunk
                    float dec = __expf(csq[mrow] - ck) * wv;
                    if (diag && (mrow < scol)) dec = 0.f;
                    Pw[m * CH + scol] = (_Float16)(sacc[t][e] * dec);
                }
            }
        }
        // Wave-local LDS RAW (D-frag layout -> A-frag layout): DS ops are
        // in-order per wave, so a dscnt drain suffices (no block barrier).
        asm volatile("s_wait_dscnt 0" ::: "memory");

        // ---- Y += P V : 2 K-steps x 4 N-tiles ----
#pragma unroll
        for (int ks = 0; ks < 2; ++ks) {
            const v16h pa = load_a16(Ps + wave * 16 * CH + ks * 32, CH, lane);
#pragma unroll
            for (int t = 0; t < 4; ++t) {
                const v16h vb = load_bT16(Vt + t * 16 * CH + ks * 32, CH, lane);
                yacc[t] = wmma_f16(pa, vb, yacc[t]);
            }
        }
    }

    // ---- store Y strip: out[b, l, h, d] f32 ----
    {
        const int nn = lane & 15;
        const int hl = (lane >> 4) & 1;
#pragma unroll
        for (int t = 0; t < 4; ++t) {
#pragma unroll
            for (int e = 0; e < 8; ++e) {
                const int lrow = l0q + wave * 16 + e + 8 * hl;
                out[((b * LL + lrow) * HH + h) * DH + t * 16 + nn] = yacc[t][e];
            }
        }
    }
}

extern "C" void kernel_launch(void* const* d_in, const int* in_sizes, int n_in,
                              void* d_out, int out_size, void* d_ws, size_t ws_size,
                              hipStream_t stream) {
    (void)in_sizes; (void)n_in; (void)out_size; (void)ws_size;
    const float* x  = (const float*)d_in[0];   // (2,2048,8,64)
    const float* dt = (const float*)d_in[1];   // (2,2048,8)
    const float* A  = (const float*)d_in[2];   // (8,)
    const float* Bp = (const float*)d_in[3];   // (2,2048,1,64)
    const float* Cp = (const float*)d_in[4];   // (2,2048,1,64)
    float* ang = (float*)d_ws;                 // 2*8*2048 f32 = 128 KB
    float* out = (float*)d_out;                // (2,2048,8,64) f32

    mamba3_scan_kernel<<<BB * HH, 256, 0, stream>>>(dt, ang);
    mamba3_attn_kernel<<<BB * HH * (LL / CH), 128, 0, stream>>>(
        x, dt, A, Bp, Cp, ang, out);
}
